// ChunkMemoryModule_43963285242295
// MI455X (gfx1250) — matrix operand order repaired and checked by reading, not compiled, not run on hardware
//
#include <hip/hip_runtime.h>
#include <hip/hip_bf16.h>

// ---------------------------------------------------------------------------
// ChunkMemoryModule on MI455X (gfx1250, wave32, WMMA, async LDS copies)
// B=4 T=4096 DIM=1024 HEADS=16 HEAD_DIM=64 CHUNK=STRIDE=64 -> C=64 chunks
// ---------------------------------------------------------------------------

typedef _Float16 v16h __attribute__((ext_vector_type(16)));
typedef _Float16 v4h  __attribute__((ext_vector_type(4)));
typedef float    v8f  __attribute__((ext_vector_type(8)));

#define GDIM 1024            // K and N of every projection GEMM
#define NT   4096            // sequence length
#define NB   4               // batch
#define NC   64              // chunks
#define NH   16              // heads
#define HD   64              // head dim

#define BM   128             // block tile M
#define BN   128             // block tile N
#define BK   32              // block tile K
#define AS   40              // padded LDS row stride (halves) for A  [m][k]
#define BS   40              // padded LDS col stride (halves) for B^T [n][k]

#define USE_ASYNC_COPY 1     // global_load_async_to_lds_b128 for A tiles

// --- CDNA5 async global->LDS copy (16B per lane), tracked by ASYNCcnt ------
__device__ __forceinline__ void async_copy16(const _Float16* gsrc,
                                             _Float16* ldst) {
#if USE_ASYNC_COPY
    unsigned loff        = (unsigned)(size_t)ldst;   // generic->LDS: addr[31:0]
    unsigned long long g = (unsigned long long)(size_t)gsrc;
    asm volatile("global_load_async_to_lds_b128 %0, %1, off"
                 :: "v"(loff), "v"(g) : "memory");
#else
    *(uint4*)ldst = *(const uint4*)gsrc;
#endif
}
__device__ __forceinline__ void wait_async() {
#if USE_ASYNC_COPY
    asm volatile("s_wait_asynccnt 0x0" ::: "memory");
#endif
}

// ---------------------------------------------------------------------------
// 0) fp32 -> fp16 conversion (one float4 per thread over n/4 float4 units)
// ---------------------------------------------------------------------------
__global__ __launch_bounds__(256)
void f32_to_f16_kernel(const float* __restrict__ src, _Float16* __restrict__ dst,
                       int n4) {
    const int i = blockIdx.x * 256 + threadIdx.x;
    if (i < n4) {
        float4 f = ((const float4*)src)[i];
        v4h h;
        h[0] = (_Float16)f.x; h[1] = (_Float16)f.y;
        h[2] = (_Float16)f.z; h[3] = (_Float16)f.w;
        ((v4h*)dst)[i] = h;
    }
}

// ---------------------------------------------------------------------------
// 1) chunk summaries from f16 x: summary[b,c,d] = mean(xh[b, start..end, d])
//    fp32 accumulation, f16 result (feeds f16 K/V projections)
// ---------------------------------------------------------------------------
__global__ __launch_bounds__(256)
void summary_kernel(const _Float16* __restrict__ xh, _Float16* __restrict__ summ) {
    const int c = blockIdx.x & 63;
    const int b = blockIdx.x >> 6;
    const int end   = c * 64;
    const int start = (end - 63) > 0 ? (end - 63) : 0;
    const float inv = 1.0f / (float)(end - start + 1);
    const int tid = threadIdx.x;

    float a0 = 0.f, a1 = 0.f, a2 = 0.f, a3 = 0.f;
    const _Float16* xp = xh + ((size_t)b * NT + start) * GDIM;
    for (int r = start; r <= end; ++r) {
        a0 += (float)xp[tid];
        a1 += (float)xp[tid + 256];
        a2 += (float)xp[tid + 512];
        a3 += (float)xp[tid + 768];
        xp += GDIM;
    }
    _Float16* sp = summ + ((size_t)b * NC + c) * GDIM;
    sp[tid]       = (_Float16)(a0 * inv);
    sp[tid + 256] = (_Float16)(a1 * inv);
    sp[tid + 512] = (_Float16)(a2 * inv);
    sp[tid + 768] = (_Float16)(a3 * inv);
}

// ---------------------------------------------------------------------------
// 2) WMMA GEMM: C[M,1024] = A[M,1024]*W[1024,1024] + bias
//    f16 operands, fp32 accumulate, OutT in {float, _Float16}.
//    Block 128x128, BK=32, 8 waves x (2x4) 16x16x32 WMMA subtiles.
//    Double-buffered LDS, ONE barrier per K-step:
//      A tiles: global->LDS via async copy (ASYNCcnt)
//      B tiles: global->regs->4x4 transpose->LDS  (Bs[n][k], contiguous K)
// ---------------------------------------------------------------------------
template <typename OutT>
__global__ __launch_bounds__(256)
void gemm_wmma_f16(const _Float16* __restrict__ A, const _Float16* __restrict__ W,
                   const float* __restrict__ bias, OutT* __restrict__ C) {
    __shared__ _Float16 As[2][BM * AS];   // [m][k], row stride AS
    __shared__ _Float16 Bs[2][BN * BS];   // [n][k] transposed, col stride BS

    const int tid  = threadIdx.x;
    const int lane = tid & 31;
    const int wave = tid >> 5;
    const int lh   = lane & 15;
    const int lsel = lane >> 4;

    const int row0 = blockIdx.y * BM;
    const int col0 = blockIdx.x * BN;
    const int wm   = (wave & 3) * 32;     // 4 waves along M
    const int wn   = (wave >> 2) * 64;    // 2 waves along N

    // A staging: 2 threads/row, 16 halves (32B = 2 async b128) each
    const int arow = tid >> 1;
    const int acol = (tid & 1) * 16;
    // B staging: thread owns 4(k) x 4(n) f16 block
    const int bk0  = (tid >> 5) * 4;
    const int bn0  = (tid & 31) * 4;

    const _Float16* agp = A + (size_t)(row0 + arow) * GDIM + acol;
    const _Float16* bgp = W + (size_t)bk0 * GDIM + col0 + bn0;

    v4h breg[4];

    // ---- prologue: stage tile 0 into buffer 0 ----
    async_copy16(agp,     &As[0][arow * AS + acol]);
    async_copy16(agp + 8, &As[0][arow * AS + acol + 8]);
    #pragma unroll
    for (int r = 0; r < 4; ++r)
        breg[r] = *(const v4h*)(bgp + (size_t)r * GDIM);
    #pragma unroll
    for (int j = 0; j < 4; ++j) {
        v4h hv; hv[0]=breg[0][j]; hv[1]=breg[1][j]; hv[2]=breg[2][j]; hv[3]=breg[3][j];
        *(v4h*)(&Bs[0][(bn0 + j) * BS + bk0]) = hv;
    }

    v8f acc[2][4] = {};
    int buf = 0;

    for (int kt = 0; kt < GDIM; kt += BK) {
        const bool more = (kt + BK) < GDIM;
        // global->regs for next B tile (overlaps previous tile's WMMAs)
        if (more) {
            const _Float16* bg = bgp + (size_t)(kt + BK) * GDIM;
            #pragma unroll
            for (int r = 0; r < 4; ++r)
                breg[r] = *(const v4h*)(bg + (size_t)r * GDIM);
        }

        wait_async();          // my async A-copies for tile kt are done
        __syncthreads();       // everyone's A+B stores for tile kt visible

        // ---- gather fragments (contiguous 32B runs -> ds_load_b128) ----
        v16h afr[2], bfr[4];
        #pragma unroll
        for (int s = 0; s < 2; ++s) {
            // A 16x32: lane = row; K = 16*lsel + i
            const _Float16* ap = &As[buf][(wm + s * 16 + lh) * AS];
            #pragma unroll
            for (int i = 0; i < 16; ++i) {
                const int j = i >> 1;
                const int k = (lsel << 4) + i;
                afr[s][i] = ap[k];
            }
        }
        #pragma unroll
        for (int s = 0; s < 4; ++s) {
            // B 32x16: lane = col; K = 16*lsel + i
            const _Float16* bp = &Bs[buf][(wn + s * 16 + lh) * BS + (lsel << 4)];
            #pragma unroll
            for (int i = 0; i < 16; ++i)
                bfr[s][i] = bp[i];
        }

        #pragma unroll
        for (int sm = 0; sm < 2; ++sm)
            #pragma unroll
            for (int sn = 0; sn < 4; ++sn)
                acc[sm][sn] = __builtin_amdgcn_wmma_f32_16x16x32_f16(
                    false, afr[sm], false, bfr[sn],
                    (short)0, acc[sm][sn], false, false);

        // ---- stage next tile into the other buffer ----
        if (more) {
            const _Float16* ag = agp + kt + BK;
            async_copy16(ag,     &As[buf ^ 1][arow * AS + acol]);
            async_copy16(ag + 8, &As[buf ^ 1][arow * AS + acol + 8]);
            #pragma unroll
            for (int j = 0; j < 4; ++j) {
                v4h hv; hv[0]=breg[0][j]; hv[1]=breg[1][j];
                        hv[2]=breg[2][j]; hv[3]=breg[3][j];
                *(v4h*)(&Bs[buf ^ 1][(bn0 + j) * BS + bk0]) = hv;
            }
        }
        buf ^= 1;
    }

    // ---- epilogue: D layout (VGPR i -> M = i + 8*lsel, N = lane&15) ----
    #pragma unroll
    for (int sm = 0; sm < 2; ++sm) {
        const int mrow = row0 + wm + sm * 16 + (lsel ? 8 : 0);
        #pragma unroll
        for (int sn = 0; sn < 4; ++sn) {
            const int ncol = col0 + wn + sn * 16 + lh;
            const float bv = bias[ncol];
            #pragma unroll
            for (int i = 0; i < 8; ++i)
                C[(size_t)(mrow + i) * GDIM + ncol] = (OutT)(acc[sm][sn][i] + bv);
        }
    }
}

// ---------------------------------------------------------------------------
// 3) chunk-causal attention, flash-style online softmax, f16 in/out,
//    fp32 math.  grid = B*H*(T/128); one query row per thread.
//    allowed chunks for position t: c <= t/64  ->  nc = t/64 + 1
// ---------------------------------------------------------------------------
__global__ __launch_bounds__(128)
void attention_kernel(const _Float16* __restrict__ Q, const _Float16* __restrict__ K,
                      const _Float16* __restrict__ V, _Float16* __restrict__ O) {
    __shared__ float ks[NC][HD];
    __shared__ float vs[NC][HD];

    const int tid  = threadIdx.x;
    const int tile = blockIdx.x & 31;          // T/128 = 32 tiles
    const int h    = (blockIdx.x >> 5) & 15;
    const int b    = blockIdx.x >> 9;

    for (int idx = tid; idx < NC * HD; idx += 128) {
        const int c = idx >> 6, d = idx & 63;
        const size_t g = ((size_t)(b * NC + c)) * GDIM + h * HD + d;
        ks[c][d] = (float)K[g];
        vs[c][d] = (float)V[g];
    }
    __syncthreads();

    const int t  = tile * 128 + tid;
    const int nc = (t >> 6) + 1;

    float q[HD];
    const _Float16* qp = Q + ((size_t)b * NT + t) * GDIM + h * HD;
    #pragma unroll
    for (int d = 0; d < HD; ++d) q[d] = (float)qp[d];

    float o[HD];
    #pragma unroll
    for (int d = 0; d < HD; ++d) o[d] = 0.0f;

    float m = -3.0e38f, l = 0.0f;
    for (int c = 0; c < nc; ++c) {
        float s = 0.0f;
        #pragma unroll
        for (int d = 0; d < HD; ++d) s += q[d] * ks[c][d];
        s *= 0.125f;                           // 1/sqrt(64)
        const float mn   = fmaxf(m, s);
        const float corr = __expf(m - mn);     // first iter: exp(-huge) = 0
        const float p    = __expf(s - mn);
        l = l * corr + p;
        #pragma unroll
        for (int d = 0; d < HD; ++d) o[d] = o[d] * corr + p * vs[c][d];
        m = mn;
    }

    const float invl = 1.0f / l;
    _Float16* op = O + ((size_t)b * NT + t) * GDIM + h * HD;
    #pragma unroll
    for (int d = 0; d < HD; ++d) op[d] = (_Float16)(o[d] * invl);
}

// ---------------------------------------------------------------------------
// launch: convert -> summary -> Q/K/V proj -> attention -> out proj
// workspace (halves): xh 16.7M | Wq/Wk/Wv/Wo h 1M each | summ 256K |
//                     Qh 16.7M | Kh 256K | Vh 256K | Abh 16.7M  ~= 111 MB
// ---------------------------------------------------------------------------
extern "C" void kernel_launch(void* const* d_in, const int* in_sizes, int n_in,
                              void* d_out, int out_size, void* d_ws, size_t ws_size,
                              hipStream_t stream) {
    const float* x  = (const float*)d_in[0];
    const float* Wq = (const float*)d_in[1];
    const float* bq = (const float*)d_in[2];
    const float* Wk = (const float*)d_in[3];
    const float* bk = (const float*)d_in[4];
    const float* Wv = (const float*)d_in[5];
    const float* bv = (const float*)d_in[6];
    const float* Wo = (const float*)d_in[7];
    const float* bo = (const float*)d_in[8];

    const size_t NXT = (size_t)NB * NT * GDIM;   // 16777216
    const size_t NW  = (size_t)GDIM * GDIM;      // 1048576
    const size_t NS  = (size_t)NB * NC * GDIM;   // 262144

    _Float16* p   = (_Float16*)d_ws;
    _Float16* xh  = p;  p += NXT;
    _Float16* Wqh = p;  p += NW;
    _Float16* Wkh = p;  p += NW;
    _Float16* Wvh = p;  p += NW;
    _Float16* Woh = p;  p += NW;
    _Float16* sm  = p;  p += NS;
    _Float16* Qh  = p;  p += NXT;
    _Float16* Kh  = p;  p += NS;
    _Float16* Vh  = p;  p += NS;
    _Float16* Abh = p;  p += NXT;

    // 0) one-time f32 -> f16 conversions
    f32_to_f16_kernel<<<dim3(NXT / 4 / 256), dim3(256), 0, stream>>>(x, xh, NXT / 4);
    f32_to_f16_kernel<<<dim3(NW / 4 / 256), dim3(256), 0, stream>>>(Wq, Wqh, NW / 4);
    f32_to_f16_kernel<<<dim3(NW / 4 / 256), dim3(256), 0, stream>>>(Wk, Wkh, NW / 4);
    f32_to_f16_kernel<<<dim3(NW / 4 / 256), dim3(256), 0, stream>>>(Wv, Wvh, NW / 4);
    f32_to_f16_kernel<<<dim3(NW / 4 / 256), dim3(256), 0, stream>>>(Wo, Woh, NW / 4);

    // 1) chunk summaries
    summary_kernel<<<dim3(NB * NC), dim3(256), 0, stream>>>(xh, sm);

    // 2) projections (WMMA GEMMs, f16 out): grid = (N/BN, M/BM)
    gemm_wmma_f16<_Float16><<<dim3(GDIM / BN, (NB * NT) / BM), dim3(256), 0, stream>>>(
        xh, Wqh, bq, Qh);
    gemm_wmma_f16<_Float16><<<dim3(GDIM / BN, (NB * NC) / BM), dim3(256), 0, stream>>>(
        sm, Wkh, bk, Kh);
    gemm_wmma_f16<_Float16><<<dim3(GDIM / BN, (NB * NC) / BM), dim3(256), 0, stream>>>(
        sm, Wvh, bv, Vh);

    // 3) chunk-causal attention
    attention_kernel<<<dim3(NB * NH * (NT / 128)), dim3(128), 0, stream>>>(
        Qh, Kh, Vh, Abh);

    // 4) output projection straight into d_out (fp32, GATING == 1.0)
    gemm_wmma_f16<float><<<dim3(GDIM / BN, (NB * NT) / BM), dim3(256), 0, stream>>>(
        Abh, Woh, bo, (float*)d_out);
}
